// Transducer_25907242730052
// MI455X (gfx1250) — matrix operand order repaired and checked by reading, not compiled, not run on hardware
//
#include <hip/hip_runtime.h>
#include <hip/hip_bf16.h>
#include <cstdint>

typedef __attribute__((ext_vector_type(16))) _Float16 v16h;
typedef __attribute__((ext_vector_type(2)))  __fp16   fp16x2;   // return type of cvt_pkrtz
typedef __attribute__((ext_vector_type(8)))  float    v8f;

static constexpr int Bn  = 4;
static constexpr int Tn  = 512;
static constexpr int Un  = 120;
static constexpr int U1n = 121;
static constexpr int Den = 320;
static constexpr int Hn  = 512;
static constexpr int Vn  = 6;
static constexpr int Rn  = 50;

// ---- branchless fast tanh: prefer the CDNA5 hardware v_tanh_f32 ----
#if __has_builtin(__builtin_amdgcn_tanhf)
__device__ __forceinline__ float fast_tanh(float x) { return __builtin_amdgcn_tanhf(x); }
#elif __has_builtin(__builtin_amdgcn_exp2f) && __has_builtin(__builtin_amdgcn_rcpf)
__device__ __forceinline__ float fast_tanh(float x) {
    // tanh(x) = 1 - 2/(exp(2x)+1), branchless via native exp2/rcp
    const float t = __builtin_amdgcn_exp2f(x * 2.885390081777927f); // 2*log2(e)
    return 1.f - 2.f * __builtin_amdgcn_rcpf(t + 1.f);
}
#else
__device__ __forceinline__ float fast_tanh(float x) { return tanhf(x); }
#endif

__device__ __forceinline__ float lae(float x, float y) {
    const float m = fmaxf(x, y);
    const float n = fminf(x, y);
    return m + __logf(1.f + __expf(n - m));   // native v_exp/v_log
}

// ---------------------------------------------------------------------------
// Kernel 1: he[row, h] = enc[row, :] @ Wf[0:320, h]        row in [0, B*T)
// ---------------------------------------------------------------------------
__global__ __launch_bounds__(256) void he_gemm(const float* __restrict__ enc,
                                               const float* __restrict__ Wf,
                                               float* __restrict__ he) {
    __shared__ float erow[Den];
    const int row = blockIdx.x;
    for (int k = threadIdx.x; k < Den; k += 256) erow[k] = enc[(size_t)row * Den + k];
    __syncthreads();
    const int h = threadIdx.x;
    float a0 = 0.f, a1 = 0.f;
    for (int k = 0; k < Den; ++k) {
        const float e = erow[k];
        a0 += e * Wf[(size_t)k * Hn + h];
        a1 += e * Wf[(size_t)k * Hn + h + 256];
    }
    he[(size_t)row * Hn + h]       = a0;
    he[(size_t)row * Hn + h + 256] = a1;
}

// ---------------------------------------------------------------------------
// Kernel 2: hd[row, h] = dec[row, :] @ Wf[320:640, h] + bf[h]   row in [0, B*U1)
// ---------------------------------------------------------------------------
__global__ __launch_bounds__(256) void hd_gemm(const float* __restrict__ dec,
                                               const float* __restrict__ Wf,
                                               const float* __restrict__ bf,
                                               float* __restrict__ hd) {
    __shared__ float drow[Den];
    const int row = blockIdx.x;
    for (int k = threadIdx.x; k < Den; k += 256) drow[k] = dec[(size_t)row * Den + k];
    __syncthreads();
    const int h = threadIdx.x;
    float a0 = bf[h], a1 = bf[h + 256];
    for (int k = 0; k < Den; ++k) {
        const float e = drow[k];
        a0 += e * Wf[(size_t)(Den + k) * Hn + h];
        a1 += e * Wf[(size_t)(Den + k) * Hn + h + 256];
    }
    hd[(size_t)row * Hn + h]       = a0;
    hd[(size_t)row * Hn + h + 256] = a1;
}

// ---------------------------------------------------------------------------
// Kernel 3: pack joint weights [512 x 64] f16 into WMMA B-fragment order.
// Col 0..5 = Wb, col 8..57 = Wr, else 0.
//   e = ((kk*4 + n)*32 + lane)*8 + j
//   k0 = kk*32 + 8*(lane>>4) + 16*(j>>2) + 2*(j&3) ; col = n*16 + (lane & 15)
// ---------------------------------------------------------------------------
__device__ __forceinline__ float wjval(const float* Wb, const float* Wr, int k, int col) {
    if (col < Vn)              return Wb[k * Vn + col];
    if (col >= 8 && col < 58)  return Wr[k * Rn + (col - 8)];
    return 0.f;
}

__global__ __launch_bounds__(256) void prep_wj(const float* __restrict__ Wb,
                                               const float* __restrict__ Wr,
                                               const float* __restrict__ bb,
                                               const float* __restrict__ br,
                                               unsigned int* __restrict__ WjSw,
                                               float* __restrict__ bias64) {
    const int idx = blockIdx.x * 256 + threadIdx.x;   // [0, 16384)
    const int j    = idx & 7;
    const int lane = (idx >> 3) & 31;
    const int n    = (idx >> 8) & 3;
    const int kk   = idx >> 10;
    const int col  = n * 16 + (lane & 15);
    const int k0   = kk * 32 + ((lane >> 4) ? 8 : 0) + 16 * (j >> 2) + 2 * (j & 3);
    union { fp16x2 h; unsigned int u; } P;
    P.h = __builtin_amdgcn_cvt_pkrtz(wjval(Wb, Wr, k0, col), wjval(Wb, Wr, k0 + 1, col));
    WjSw[idx] = P.u;
    if (idx < 64) {
        float v = 0.f;
        if (idx < Vn)                  v = bb[idx];
        else if (idx >= 8 && idx < 58) v = br[idx - 8];
        bias64[idx] = v;
    }
}

// ---------------------------------------------------------------------------
// Kernel 4 (fused joint): per (b, t, half), 4 waves x 16 u-rows each.
// h = tanh(he + hd') -> f16 A fragments -> 64x v_wmma_f32_16x16x32_f16 per
// wave -> logits via LDS -> per-row logsumexp -> 4 floats per (b,t,u).
// ---------------------------------------------------------------------------
__global__ __launch_bounds__(128) void joint_fused(
    const float* __restrict__ he, const float* __restrict__ hd,
    const unsigned int* __restrict__ WjSw, const float* __restrict__ bias64,
    const int* __restrict__ targets, const int* __restrict__ rles,
    float* __restrict__ blankB, float* __restrict__ emitB,
    float* __restrict__ blankR, float* __restrict__ emitR) {

    __shared__ float he_sh[Hn];
    __shared__ float logits[4][16 * 64];

    const int bid  = blockIdx.x;            // [0, B*T*2)
    const int half = bid & 1;
    const int t    = (bid >> 1) & (Tn - 1);
    const int b    = bid >> 10;             // / (T*2)

    const float* herow = he + ((size_t)(b * Tn + t)) * Hn;
    for (int k = threadIdx.x; k < Hn; k += 128) he_sh[k] = herow[k];
    __syncthreads();

    const int lane = threadIdx.x & 31;
    const int wave = threadIdx.x >> 5;
    const int u0   = half * 64 + wave * 16;
    const int hi   = lane >> 4;
    const int arow = lane & 15;                   // A-matrix row owned by this lane
    const int ua   = (u0 + arow > Un) ? Un : (u0 + arow);
    const float* hdrow = hd + ((size_t)(b * U1n + ua)) * Hn;

    v8f c0 = {}, c1 = {}, c2 = {}, c3 = {};
    const uint4* wq = (const uint4*)WjSw;

    for (int kk = 0; kk < 16; ++kk) {
        const int kb = kk * 32 + hi * 8;
        const float4 d0 = *(const float4*)(hdrow + kb + 0);
        const float4 d1 = *(const float4*)(hdrow + kb + 4);
        const float4 d2 = *(const float4*)(hdrow + kb + 16);
        const float4 d3 = *(const float4*)(hdrow + kb + 20);
        const float4 e0 = *(const float4*)(he_sh + kb + 0);
        const float4 e1 = *(const float4*)(he_sh + kb + 4);
        const float4 e2 = *(const float4*)(he_sh + kb + 16);
        const float4 e3 = *(const float4*)(he_sh + kb + 20);

        union AFrag { fp16x2 p[8]; v16h v; } A;
        A.p[0] = __builtin_amdgcn_cvt_pkrtz(fast_tanh(d0.x + e0.x), fast_tanh(d0.y + e0.y));
        A.p[1] = __builtin_amdgcn_cvt_pkrtz(fast_tanh(d0.z + e0.z), fast_tanh(d0.w + e0.w));
        A.p[2] = __builtin_amdgcn_cvt_pkrtz(fast_tanh(d1.x + e1.x), fast_tanh(d1.y + e1.y));
        A.p[3] = __builtin_amdgcn_cvt_pkrtz(fast_tanh(d1.z + e1.z), fast_tanh(d1.w + e1.w));
        A.p[4] = __builtin_amdgcn_cvt_pkrtz(fast_tanh(d2.x + e2.x), fast_tanh(d2.y + e2.y));
        A.p[5] = __builtin_amdgcn_cvt_pkrtz(fast_tanh(d2.z + e2.z), fast_tanh(d2.w + e2.w));
        A.p[6] = __builtin_amdgcn_cvt_pkrtz(fast_tanh(d3.x + e3.x), fast_tanh(d3.y + e3.y));
        A.p[7] = __builtin_amdgcn_cvt_pkrtz(fast_tanh(d3.z + e3.z), fast_tanh(d3.w + e3.w));

        union BFrag { uint4 q[2]; v16h h; } b0, b1, b2, b3;
        const int base = (kk * 4) * 32 + lane;            // entry index; x2 for uint4
        b0.q[0] = wq[(base + 0 * 32) * 2 + 0]; b0.q[1] = wq[(base + 0 * 32) * 2 + 1];
        b1.q[0] = wq[(base + 1 * 32) * 2 + 0]; b1.q[1] = wq[(base + 1 * 32) * 2 + 1];
        b2.q[0] = wq[(base + 2 * 32) * 2 + 0]; b2.q[1] = wq[(base + 2 * 32) * 2 + 1];
        b3.q[0] = wq[(base + 3 * 32) * 2 + 0]; b3.q[1] = wq[(base + 3 * 32) * 2 + 1];

        c0 = __builtin_amdgcn_wmma_f32_16x16x32_f16(false, A.v, false, b0.h, (short)0, c0, false, false);
        c1 = __builtin_amdgcn_wmma_f32_16x16x32_f16(false, A.v, false, b1.h, (short)0, c1, false, false);
        c2 = __builtin_amdgcn_wmma_f32_16x16x32_f16(false, A.v, false, b2.h, (short)0, c2, false, false);
        c3 = __builtin_amdgcn_wmma_f32_16x16x32_f16(false, A.v, false, b3.h, (short)0, c3, false, false);
    }

    // C/D layout: lane holds column (lane&15) of each tile, rows r + 8*hi.
    float* lg = logits[wave];
    const int colt = lane & 15;
#pragma unroll
    for (int r = 0; r < 8; ++r) {
        const int mm = r + 8 * hi;
        lg[mm * 64 + 0  + colt] = c0[r];
        lg[mm * 64 + 16 + colt] = c1[r];
        lg[mm * 64 + 32 + colt] = c2[r];
        lg[mm * 64 + 48 + colt] = c3[r];
    }
    __syncthreads();

    if (lane < 16) {
        const int uu = u0 + lane;
        if (uu <= Un) {
            const float* z = logits[wave] + lane * 64;
            const size_t oidx = ((size_t)(b * Tn + t)) * U1n + uu;
            // ---- base head: cols 0..5 ----
            float zb[Vn];
            float mb = -1e30f;
#pragma unroll
            for (int c = 0; c < Vn; ++c) { zb[c] = z[c] + bias64[c]; mb = fmaxf(mb, zb[c]); }
            float sb = 0.f;
#pragma unroll
            for (int c = 0; c < Vn; ++c) sb += __expf(zb[c] - mb);
            const float lseb = mb + __logf(sb);
            blankB[oidx] = zb[0] - lseb;
            const int tg = (uu < Un) ? targets[b * Un + uu] : 0;
            emitB[oidx]  = zb[tg] - lseb;
            // ---- rle head: cols 8..57 ----
            float mr = -1e30f;
            for (int c = 0; c < Rn; ++c) mr = fmaxf(mr, z[8 + c] + bias64[8 + c]);
            float sr = 0.f;
            for (int c = 0; c < Rn; ++c) sr += __expf(z[8 + c] + bias64[8 + c] - mr);
            const float lser = mr + __logf(sr);
            blankR[oidx] = (z[8] + bias64[8]) - lser;
            const int rg = (uu < Un) ? rles[b * Un + uu] : 0;
            emitR[oidx]  = (z[8 + rg] + bias64[8 + rg]) - lser;
        }
    }
}

// ---------------------------------------------------------------------------
// Kernel 5: RNNT forward DP, anti-diagonal wavefront. 8 blocks = {base,rle}xB.
// ---------------------------------------------------------------------------
__global__ __launch_bounds__(128) void rnnt_dp(
    const float* __restrict__ blankB, const float* __restrict__ emitB,
    const float* __restrict__ blankR, const float* __restrict__ emitR,
    const int* __restrict__ tlen, const int* __restrict__ ulen,
    float* __restrict__ ll) {

    const int b    = blockIdx.x & 3;
    const int task = blockIdx.x >> 2;
    const float* Bp = (task ? blankR : blankB) + (size_t)b * Tn * U1n;
    const float* Ep = (task ? emitR  : emitB)  + (size_t)b * Tn * U1n;
    const int TL = tlen[b];
    const int UL = ulen[b];

    __shared__ float dA[U1n];
    __shared__ float dB[U1n];
    float* prev = dA;
    float* cur  = dB;

    const int u = threadIdx.x;
    for (int d = 0; d <= (Tn - 1) + Un; ++d) {
        const int t = d - u;
        if (u <= Un && t >= 0 && t < Tn) {
            float a;
            if (d == 0)      a = 0.f;
            else if (u == 0) a = prev[0]     + Bp[(size_t)(t - 1) * U1n];
            else if (t == 0) a = prev[u - 1] + Ep[(size_t)(u - 1)];
            else {
                const float x = prev[u]     + Bp[(size_t)(t - 1) * U1n + u];
                const float y = prev[u - 1] + Ep[(size_t)t * U1n + (u - 1)];
                a = lae(x, y);
            }
            cur[u] = a;
            if (t == TL - 1 && u == UL)
                ll[blockIdx.x] = a + Bp[(size_t)t * U1n + u];
        }
        __syncthreads();
        float* tmp = prev; prev = cur; cur = tmp;
    }
}

// ---------------------------------------------------------------------------
// Kernel 6: combine -> scalar loss
// ---------------------------------------------------------------------------
__global__ void finalize(const float* __restrict__ ll, float* __restrict__ out) {
    if (threadIdx.x == 0) {
        float sb = 0.f, sr = 0.f;
        for (int i = 0; i < Bn; ++i) { sb += ll[i]; sr += ll[4 + i]; }
        out[0] = -(sb * 0.25f) - 0.1f * (sr * 0.25f);
    }
}

// ---------------------------------------------------------------------------
extern "C" void kernel_launch(void* const* d_in, const int* in_sizes, int n_in,
                              void* d_out, int out_size, void* d_ws, size_t ws_size,
                              hipStream_t stream) {
    const float* enc = (const float*)d_in[0];
    const float* dec = (const float*)d_in[1];
    const float* Wf  = (const float*)d_in[2];
    const float* bf  = (const float*)d_in[3];
    const float* Wb  = (const float*)d_in[4];
    const float* bb  = (const float*)d_in[5];
    const float* Wr  = (const float*)d_in[6];
    const float* br  = (const float*)d_in[7];
    const int*   targets = (const int*)d_in[8];
    const int*   rles    = (const int*)d_in[9];
    const int*   ilen    = (const int*)d_in[10];
    const int*   tgl     = (const int*)d_in[11];
    float* out = (float*)d_out;

    char* ws = (char*)d_ws;
    size_t off = 0;
    auto alloc = [&](size_t bytes) { size_t o = off; off = (off + bytes + 255) & ~(size_t)255; return o; };
    const size_t planeB = (size_t)Bn * Tn * U1n * sizeof(float);

    float*        he_p   = (float*)(ws + alloc((size_t)Bn * Tn * Hn * sizeof(float)));
    float*        hd_p   = (float*)(ws + alloc((size_t)Bn * U1n * Hn * sizeof(float)));
    unsigned int* WjSw   = (unsigned int*)(ws + alloc(16384 * sizeof(unsigned int)));
    float*        bias64 = (float*)(ws + alloc(64 * sizeof(float)));
    float*        blankB = (float*)(ws + alloc(planeB));
    float*        emitB  = (float*)(ws + alloc(planeB));
    float*        blankR = (float*)(ws + alloc(planeB));
    float*        emitR  = (float*)(ws + alloc(planeB));
    float*        llp    = (float*)(ws + alloc(8 * sizeof(float)));

    he_gemm<<<Bn * Tn, 256, 0, stream>>>(enc, Wf, he_p);
    hd_gemm<<<Bn * U1n, 256, 0, stream>>>(dec, Wf, bf, hd_p);
    prep_wj<<<64, 256, 0, stream>>>(Wb, Wr, bb, br, WjSw, bias64);
    joint_fused<<<Bn * Tn * 2, 128, 0, stream>>>(he_p, hd_p, WjSw, bias64,
                                                 targets, rles,
                                                 blankB, emitB, blankR, emitR);
    rnnt_dp<<<8, 128, 0, stream>>>(blankB, emitB, blankR, emitR, ilen, tgl, llp);
    finalize<<<1, 32, 0, stream>>>(llp, out);
}
